// TensorProductScoreModel_11347303596714
// MI455X (gfx1250) — matrix operand order repaired and checked by reading, not compiled, not run on hardware
//
#include <hip/hip_runtime.h>
#include <hip/hip_bf16.h>
#include <math.h>

typedef __attribute__((ext_vector_type(16))) _Float16 v16h;
typedef __attribute__((ext_vector_type(8)))  _Float16 v8h;
typedef __attribute__((ext_vector_type(8)))  float    v8f;

#define N_NODES_C 100000
#define N_EDGES_C 400000
#define HSTRIDE   192
#define ME        32   // M rows (edges/nodes) per block: 2 WMMA subtiles

// ---------------------------------------------------------------------------
// WMMA helpers (gfx1250, wave32). D = A(16x32 f16) x B(32x16 f16) + C(16x16 f32)
// ---------------------------------------------------------------------------
__device__ __forceinline__ v8f wmma32(v16h a, v16h b, v8f c) {
  return __builtin_amdgcn_wmma_f32_16x16x32_f16(false, a, false, b, (short)0, c,
                                                false, false);
}

// A fragment from row-major f16 LDS tile: row = lane&15,
// lanes 0-15 hold K = k0..k0+15, lanes 16-31 K = k0+16..k0+31.
__device__ __forceinline__ v16h fragA_lds(const _Float16* S, int ld, int k0, int lane) {
  const _Float16* p = S + (lane & 15) * ld + k0 + ((lane & 16) ? 16 : 0);
  v8h lo = *(const v8h*)p;
  v8h hi = *(const v8h*)(p + 8);
  v16h a;
#pragma unroll
  for (int j = 0; j < 8; ++j) { a[j] = lo[j]; a[8 + j] = hi[j]; }
  return a;
}

// B fragment from pre-packed transposed f16 weights Wt[N][Kpad]:
// col = n0+(lane&15); lanes 0-15 hold K=k0..k0+15, lanes 16-31 K=k0+16..k0+31.
__device__ __forceinline__ v16h fragB_t(const _Float16* Wt, int ldk, int k0, int n0,
                                        int lane) {
  int n  = n0 + (lane & 15);
  int kb = k0 + ((lane & 16) ? 16 : 0);
  return *(const v16h*)(Wt + (size_t)n * ldk + kb);
}

// ---------------------------------------------------------------------------
// Weight packing: W[K x N] f32 row-major -> Wt[Npad][Kpad] f16, zero padded.
// ---------------------------------------------------------------------------
__global__ void k_pack(const float* __restrict__ W, _Float16* __restrict__ Wt,
                       int K, int N, int Kpad, int Npad) {
  int i = blockIdx.x * blockDim.x + threadIdx.x;
  if (i >= Npad * Kpad) return;
  int n = i / Kpad, k = i - n * Kpad;
  float v = (k < K && n < N) ? W[(size_t)k * N + n] : 0.0f;
  Wt[i] = (_Float16)v;
}

// ---------------------------------------------------------------------------
// Time embedding: temb[g][0:16]=sin(t*f_j), [16:32]=cos(t*f_j)
// ---------------------------------------------------------------------------
__global__ void k_temb(const int* __restrict__ timeI, float* __restrict__ temb) {
  int i = blockIdx.x * blockDim.x + threadIdx.x;
  if (i >= 256 * 32) return;
  int g = i >> 5, d = i & 31;
  int j = d & 15;
  float freq = __expf(-logf(10000.0f) * (float)j / 16.0f);
  float ang = (float)timeI[g] * freq;
  temb[i] = (d < 16) ? __sinf(ang) : __cosf(ang);
}

// ---------------------------------------------------------------------------
// In-degree counts (same for every layer)
// ---------------------------------------------------------------------------
__global__ void k_counts(const int* __restrict__ dstI, float* __restrict__ cnt, int nE) {
  int i = blockIdx.x * blockDim.x + threadIdx.x;
  if (i < nE) atomicAdd(&cnt[dstI[i]], 1.0f);
}

// ---------------------------------------------------------------------------
// Node MLP: h0 = relu(concat(x,temb[batch]) @ W1 + b1) @ W2 + b2   (106->32->32)
// 32 nodes / block, 2 waves; each wave one N-tile across 2 M-subtiles.
// ---------------------------------------------------------------------------
__global__ __launch_bounds__(64) void k_node_mlp(
    const float* __restrict__ x, const int* __restrict__ batch,
    const float* __restrict__ temb,
    const _Float16* __restrict__ w1t, const float* __restrict__ b1,
    const _Float16* __restrict__ w2t, const float* __restrict__ b2,
    float* __restrict__ hout) {
  __shared__ _Float16 s_na[ME][128];
  __shared__ _Float16 s_hid[ME][32];
  int n0 = blockIdx.x * ME;
  int tid = threadIdx.x, lane = tid & 31, wv = tid >> 5;

  for (int i = tid; i < ME * 128; i += 64) {
    int m = i >> 7, c = i & 127;
    int node = n0 + m;
    float v = 0.0f;
    if (c < 74)       v = x[(size_t)node * 74 + c];
    else if (c < 106) v = temb[batch[node] * 32 + (c - 74)];
    s_na[m][c] = (_Float16)v;
  }
  __syncthreads();

  int mh = (lane >> 4) * 8;
  int nl = lane & 15;
  int n = wv * 16 + nl;
  { // GEMM1: K=128 (padded), N-tile = wv, 2 M-subtiles share each B frag
    v8f a0 = {0, 0, 0, 0, 0, 0, 0, 0}, a1 = {0, 0, 0, 0, 0, 0, 0, 0};
#pragma unroll
    for (int kc = 0; kc < 4; ++kc) {
      v16h b  = fragB_t(w1t, 128, kc * 32, wv * 16, lane);
      v16h f0 = fragA_lds(&s_na[0][0],  128, kc * 32, lane);
      v16h f1 = fragA_lds(&s_na[16][0], 128, kc * 32, lane);
      a0 = wmma32(f0, b, a0);
      a1 = wmma32(f1, b, a1);
    }
    float bias = b1[n];
#pragma unroll
    for (int v = 0; v < 8; ++v) {
      float r0 = a0[v] + bias, r1 = a1[v] + bias;
      s_hid[mh + v][n]      = (_Float16)(r0 > 0.0f ? r0 : 0.0f);
      s_hid[16 + mh + v][n] = (_Float16)(r1 > 0.0f ? r1 : 0.0f);
    }
  }
  __syncthreads();
  { // GEMM2: K=32
    v8f a0 = {0, 0, 0, 0, 0, 0, 0, 0}, a1 = {0, 0, 0, 0, 0, 0, 0, 0};
    v16h b  = fragB_t(w2t, 32, 0, wv * 16, lane);
    v16h f0 = fragA_lds(&s_hid[0][0],  32, 0, lane);
    v16h f1 = fragA_lds(&s_hid[16][0], 32, 0, lane);
    a0 = wmma32(f0, b, a0);
    a1 = wmma32(f1, b, a1);
    float bias = b2[n];
#pragma unroll
    for (int v = 0; v < 8; ++v) {
      hout[(size_t)(n0 + mh + v) * HSTRIDE + n]      = a0[v] + bias;
      hout[(size_t)(n0 + 16 + mh + v) * HSTRIDE + n] = a1[v] + bias;
    }
  }
}

// ---------------------------------------------------------------------------
// Edge featurization + MLP (86->32->32) + spherical harmonics
// 32 edges / block, 2 waves.
// ---------------------------------------------------------------------------
__global__ __launch_bounds__(64) void k_edge_mlp(
    const float* __restrict__ pos, const float* __restrict__ eattr,
    const int* __restrict__ srcI, const int* __restrict__ dstI,
    const int* __restrict__ batch, const float* __restrict__ temb,
    const _Float16* __restrict__ w1t, const float* __restrict__ b1,
    const _Float16* __restrict__ w2t, const float* __restrict__ b2,
    float* __restrict__ efeat, float* __restrict__ shout) {
  __shared__ _Float16 s_ea[ME][96];
  __shared__ _Float16 s_hid[ME][32];
  int e0 = blockIdx.x * ME;
  int tid = threadIdx.x, lane = tid & 31, wv = tid >> 5;

  if (tid < ME) {
    int e = e0 + tid;
    int s = srcI[e], d = dstI[e];
    float ex = pos[d * 3 + 0] - pos[s * 3 + 0];
    float ey = pos[d * 3 + 1] - pos[s * 3 + 1];
    float ez = pos[d * 3 + 2] - pos[s * 3 + 2];
    float dn = sqrtf(ex * ex + ey * ey + ez * ez);
    float inv = 1.0f / (dn + 1e-8f);
    float nx = ex * inv, ny = ey * inv, nz = ez * inv;
    const float s3 = 1.7320508075688772f, s5 = 2.23606797749979f,
                s15 = 3.872983346207417f;
    float sh[9];
    sh[0] = 1.0f;           sh[1] = s3 * nx;        sh[2] = s3 * ny;
    sh[3] = s3 * nz;        sh[4] = s15 * nx * ny;  sh[5] = s15 * ny * nz;
    sh[6] = 0.5f * s5 * (3.0f * nz * nz - 1.0f);
    sh[7] = s15 * nx * nz;  sh[8] = 0.5f * s15 * (nx * nx - ny * ny);
#pragma unroll
    for (int j = 0; j < 9; ++j) shout[(size_t)e * 9 + j] = sh[j];
#pragma unroll
    for (int c = 0; c < 4; ++c) s_ea[tid][c] = (_Float16)eattr[(size_t)e * 4 + c];
    int g = batch[s];
    for (int c = 0; c < 32; ++c) s_ea[tid][4 + c] = (_Float16)temb[g * 32 + c];
    const float delta = 5.0f / 49.0f;
    const float coeff = -0.5f / (delta * delta);
    for (int j = 0; j < 50; ++j) {
      float diff = dn - (float)j * delta;
      s_ea[tid][36 + j] = (_Float16)__expf(coeff * diff * diff);
    }
    for (int c = 86; c < 96; ++c) s_ea[tid][c] = (_Float16)0.0f;
  }
  __syncthreads();

  int mh = (lane >> 4) * 8;
  int nl = lane & 15;
  int n = wv * 16 + nl;
  { // GEMM1: K=96 (padded)
    v8f a0 = {0, 0, 0, 0, 0, 0, 0, 0}, a1 = {0, 0, 0, 0, 0, 0, 0, 0};
#pragma unroll
    for (int kc = 0; kc < 3; ++kc) {
      v16h b  = fragB_t(w1t, 96, kc * 32, wv * 16, lane);
      v16h f0 = fragA_lds(&s_ea[0][0],  96, kc * 32, lane);
      v16h f1 = fragA_lds(&s_ea[16][0], 96, kc * 32, lane);
      a0 = wmma32(f0, b, a0);
      a1 = wmma32(f1, b, a1);
    }
    float bias = b1[n];
#pragma unroll
    for (int v = 0; v < 8; ++v) {
      float r0 = a0[v] + bias, r1 = a1[v] + bias;
      s_hid[mh + v][n]      = (_Float16)(r0 > 0.0f ? r0 : 0.0f);
      s_hid[16 + mh + v][n] = (_Float16)(r1 > 0.0f ? r1 : 0.0f);
    }
  }
  __syncthreads();
  { // GEMM2: K=32
    v8f a0 = {0, 0, 0, 0, 0, 0, 0, 0}, a1 = {0, 0, 0, 0, 0, 0, 0, 0};
    v16h b  = fragB_t(w2t, 32, 0, wv * 16, lane);
    v16h f0 = fragA_lds(&s_hid[0][0],  32, 0, lane);
    v16h f1 = fragA_lds(&s_hid[16][0], 32, 0, lane);
    a0 = wmma32(f0, b, a0);
    a1 = wmma32(f1, b, a1);
    float bias = b2[n];
#pragma unroll
    for (int v = 0; v < 8; ++v) {
      efeat[(size_t)(e0 + mh + v) * 32 + n]      = a0[v] + bias;
      efeat[(size_t)(e0 + 16 + mh + v) * 32 + n] = a1[v] + bias;
    }
  }
}

// ---------------------------------------------------------------------------
// Fused interaction layer: 32 edges / block (2 WMMA subtiles), 4 waves.
//   es(96) -> relu MLP -> w(288, f16 in LDS);  xs = h_src @ A;
//   tp computed in-place on s_w;  msg = tp @ B;  atomic scatter to accum[dst].
// ---------------------------------------------------------------------------
__global__ __launch_bounds__(128) void k_layer(
    const float* __restrict__ h, const float* __restrict__ efeat,
    const float* __restrict__ shg,
    const int* __restrict__ srcI, const int* __restrict__ dstI,
    const _Float16* __restrict__ At,
    const _Float16* __restrict__ fw1t, const float* __restrict__ fcb1,
    const _Float16* __restrict__ fw2t, const float* __restrict__ fcb2,
    const _Float16* __restrict__ Bt,
    float* __restrict__ accum, int din, int dout) {
  __shared__ _Float16 s_es[ME][96];
  __shared__ _Float16 s_hs[ME][HSTRIDE];
  __shared__ _Float16 s_hid[ME][96];
  __shared__ _Float16 s_w[ME][288];   // fc2 output, then tensor-product in place
  __shared__ float    s_sh[ME][9];
  __shared__ float    s_xs[ME][32];
  __shared__ int      s_src[ME];
  __shared__ int      s_dst[ME];

  int e0 = blockIdx.x * ME;
  int tid = threadIdx.x, lane = tid & 31, wv = tid >> 5;
  int mh = (lane >> 4) * 8;
  int nl = lane & 15;

  if (tid < ME) { s_src[tid] = srcI[e0 + tid]; s_dst[tid] = dstI[e0 + tid]; }
  __syncthreads();

  // Gather. s_hs columns >= din are never consumed (xs GEMM runs K=din only).
  for (int i = tid; i < ME * 96; i += 128) {
    int m = i / 96, c = i % 96;
    float v;
    if (c < 32)      v = efeat[(size_t)(e0 + m) * 32 + c];
    else if (c < 64) v = h[(size_t)s_src[m] * HSTRIDE + (c - 32)];
    else             v = h[(size_t)s_dst[m] * HSTRIDE + (c - 64)];
    s_es[m][c] = (_Float16)v;
  }
  for (int i = tid; i < ME * HSTRIDE; i += 128) {
    int m = i / HSTRIDE, c = i % HSTRIDE;
    s_hs[m][c] = (_Float16)h[(size_t)s_src[m] * HSTRIDE + c];
  }
  for (int i = tid; i < ME * 9; i += 128)
    s_sh[i / 9][i % 9] = shg[(size_t)e0 * 9 + i];
  __syncthreads();

  // GEMM1: hidden = relu(es @ fcw1 + fcb1), 6 N-tiles, K=96
  for (int nt = wv; nt < 6; nt += 4) {
    v8f a0 = {0, 0, 0, 0, 0, 0, 0, 0}, a1 = {0, 0, 0, 0, 0, 0, 0, 0};
#pragma unroll
    for (int kc = 0; kc < 3; ++kc) {
      v16h b  = fragB_t(fw1t, 96, kc * 32, nt * 16, lane);
      v16h f0 = fragA_lds(&s_es[0][0],  96, kc * 32, lane);
      v16h f1 = fragA_lds(&s_es[16][0], 96, kc * 32, lane);
      a0 = wmma32(f0, b, a0);
      a1 = wmma32(f1, b, a1);
    }
    int n = nt * 16 + nl;
    float bias = fcb1[n];
#pragma unroll
    for (int v = 0; v < 8; ++v) {
      float r0 = a0[v] + bias, r1 = a1[v] + bias;
      s_hid[mh + v][n]      = (_Float16)(r0 > 0.0f ? r0 : 0.0f);
      s_hid[16 + mh + v][n] = (_Float16)(r1 > 0.0f ? r1 : 0.0f);
    }
  }
  __syncthreads();

  // GEMM2: w = hidden @ fcw2 + fcb2 (18 tiles) and xs = h_src @ A (2 tiles)
  int kchA = din >> 5;
  for (int vt = wv; vt < 20; vt += 4) {
    if (vt < 18) {
      v8f a0 = {0, 0, 0, 0, 0, 0, 0, 0}, a1 = {0, 0, 0, 0, 0, 0, 0, 0};
#pragma unroll
      for (int kc = 0; kc < 3; ++kc) {
        v16h b  = fragB_t(fw2t, 96, kc * 32, vt * 16, lane);
        v16h f0 = fragA_lds(&s_hid[0][0],  96, kc * 32, lane);
        v16h f1 = fragA_lds(&s_hid[16][0], 96, kc * 32, lane);
        a0 = wmma32(f0, b, a0);
        a1 = wmma32(f1, b, a1);
      }
      int n = vt * 16 + nl;
      float bias = fcb2[n];
#pragma unroll
      for (int v = 0; v < 8; ++v) {
        s_w[mh + v][n]      = (_Float16)(a0[v] + bias);
        s_w[16 + mh + v][n] = (_Float16)(a1[v] + bias);
      }
    } else {
      int nt = vt - 18;
      v8f a0 = {0, 0, 0, 0, 0, 0, 0, 0}, a1 = {0, 0, 0, 0, 0, 0, 0, 0};
      for (int kc = 0; kc < kchA; ++kc) {
        v16h b  = fragB_t(At, din, kc * 32, nt * 16, lane);
        v16h f0 = fragA_lds(&s_hs[0][0],  HSTRIDE, kc * 32, lane);
        v16h f1 = fragA_lds(&s_hs[16][0], HSTRIDE, kc * 32, lane);
        a0 = wmma32(f0, b, a0);
        a1 = wmma32(f1, b, a1);
      }
      int n = nt * 16 + nl;
#pragma unroll
      for (int v = 0; v < 8; ++v) {
        s_xs[mh + v][n]      = a0[v];
        s_xs[16 + mh + v][n] = a1[v];
      }
    }
  }
  __syncthreads();

  // Tensor product in place: w[m][r*9+s] *= xs[m][r]*sh[m][s]
  for (int i = tid; i < ME * 288; i += 128) {
    int m = i / 288, c = i % 288;
    int r = c / 9, s = c - r * 9;
    s_w[m][c] = (_Float16)((float)s_w[m][c] * s_xs[m][r] * s_sh[m][s]);
  }
  __syncthreads();

  // GEMM3: msg = tp @ B (K=288, N=dout), scatter-add to accum[dst]
  int ntiles = dout >> 4;
  for (int nt = wv; nt < ntiles; nt += 4) {
    v8f a0 = {0, 0, 0, 0, 0, 0, 0, 0}, a1 = {0, 0, 0, 0, 0, 0, 0, 0};
#pragma unroll
    for (int kc = 0; kc < 9; ++kc) {
      v16h b  = fragB_t(Bt, 288, kc * 32, nt * 16, lane);
      v16h f0 = fragA_lds(&s_w[0][0],  288, kc * 32, lane);
      v16h f1 = fragA_lds(&s_w[16][0], 288, kc * 32, lane);
      a0 = wmma32(f0, b, a0);
      a1 = wmma32(f1, b, a1);
    }
    int n = nt * 16 + nl;
#pragma unroll
    for (int v = 0; v < 8; ++v) {
      atomicAdd(&accum[(size_t)s_dst[mh + v] * HSTRIDE + n], a0[v]);
      atomicAdd(&accum[(size_t)s_dst[16 + mh + v] * HSTRIDE + n], a1[v]);
    }
  }
}

// ---------------------------------------------------------------------------
// Finalize: h_next = accum/max(cnt,1) + pad(h_prev)   (in-place on accum)
// ---------------------------------------------------------------------------
__global__ void k_finalize(float* __restrict__ accum, const float* __restrict__ hprev,
                           const float* __restrict__ cnt, int din, int dout, int nN) {
  int i = blockIdx.x * blockDim.x + threadIdx.x;
  if (i >= nN * dout) return;
  int n = i / dout, c = i - n * dout;
  float cc = cnt[n];
  if (cc < 1.0f) cc = 1.0f;
  float v = accum[(size_t)n * HSTRIDE + c] / cc;
  if (c < din) v += hprev[(size_t)n * HSTRIDE + c];
  accum[(size_t)n * HSTRIDE + c] = v;
}

// ---------------------------------------------------------------------------
// Readout: out = tanh(h @ ro_w1 + ro_b1) @ ro_w2   (192->96->3), 32 nodes/block
// ---------------------------------------------------------------------------
__global__ __launch_bounds__(128) void k_readout(
    const float* __restrict__ h,
    const _Float16* __restrict__ w1t, const float* __restrict__ b1,
    const _Float16* __restrict__ w2t, float* __restrict__ out) {
  __shared__ _Float16 s_h[ME][HSTRIDE];
  __shared__ _Float16 s_hid[ME][96];
  int n0 = blockIdx.x * ME;
  int tid = threadIdx.x, lane = tid & 31, wv = tid >> 5;
  int mh = (lane >> 4) * 8;
  int nl = lane & 15;

  for (int i = tid; i < ME * HSTRIDE; i += 128) {
    int m = i / HSTRIDE, c = i % HSTRIDE;
    s_h[m][c] = (_Float16)h[(size_t)(n0 + m) * HSTRIDE + c];
  }
  __syncthreads();

  for (int nt = wv; nt < 6; nt += 4) {
    v8f a0 = {0, 0, 0, 0, 0, 0, 0, 0}, a1 = {0, 0, 0, 0, 0, 0, 0, 0};
#pragma unroll
    for (int kc = 0; kc < 6; ++kc) {
      v16h b  = fragB_t(w1t, 192, kc * 32, nt * 16, lane);
      v16h f0 = fragA_lds(&s_h[0][0],  HSTRIDE, kc * 32, lane);
      v16h f1 = fragA_lds(&s_h[16][0], HSTRIDE, kc * 32, lane);
      a0 = wmma32(f0, b, a0);
      a1 = wmma32(f1, b, a1);
    }
    int n = nt * 16 + nl;
    float bias = b1[n];
#pragma unroll
    for (int v = 0; v < 8; ++v) {
      s_hid[mh + v][n]      = (_Float16)tanhf(a0[v] + bias);
      s_hid[16 + mh + v][n] = (_Float16)tanhf(a1[v] + bias);
    }
  }
  __syncthreads();

  if (wv == 0) { // final 96->3 (N padded to 16 in packed weights)
    v8f a0 = {0, 0, 0, 0, 0, 0, 0, 0}, a1 = {0, 0, 0, 0, 0, 0, 0, 0};
#pragma unroll
    for (int kc = 0; kc < 3; ++kc) {
      v16h b  = fragB_t(w2t, 96, kc * 32, 0, lane);
      v16h f0 = fragA_lds(&s_hid[0][0],  96, kc * 32, lane);
      v16h f1 = fragA_lds(&s_hid[16][0], 96, kc * 32, lane);
      a0 = wmma32(f0, b, a0);
      a1 = wmma32(f1, b, a1);
    }
    if (nl < 3) {
#pragma unroll
      for (int v = 0; v < 8; ++v) {
        out[(size_t)(n0 + mh + v) * 3 + nl]      = a0[v];
        out[(size_t)(n0 + 16 + mh + v) * 3 + nl] = a1[v];
      }
    }
  }
}

// ---------------------------------------------------------------------------
// Host orchestration
// ---------------------------------------------------------------------------
extern "C" void kernel_launch(void* const* d_in, const int* in_sizes, int n_in,
                              void* d_out, int out_size, void* d_ws, size_t ws_size,
                              hipStream_t stream) {
  (void)out_size; (void)ws_size;
  const float *x = nullptr, *pos = nullptr, *eattr = nullptr;
  const int *eidx = nullptr, *batch = nullptr, *timeI = nullptr;
  const float* leaf[64];
  int lsz[64];
  int nl = 0;

  for (int i = 0; i < n_in; ++i) {
    switch (in_sizes[i]) {
      case 7400000: x = (const float*)d_in[i]; break;
      case 300000:  pos = (const float*)d_in[i]; break;
      case 1600000: eattr = (const float*)d_in[i]; break;
      case 800000:  eidx = (const int*)d_in[i]; break;
      case 100000:  batch = (const int*)d_in[i]; break;
      case 256:     timeI = (const int*)d_in[i]; break;
      default:
        if (nl < 64) { leaf[nl] = (const float*)d_in[i]; lsz[nl] = in_sizes[i]; ++nl; }
        break;
    }
  }
  if (!x || !pos || !eattr || !eidx || !batch || !timeI || nl != 35) return;

  // Param leaf ordering detection: insertion-order vs jax sorted-key flatten.
  const float *node_w1, *node_b1, *node_w2, *node_b2;
  const float *edge_w1, *edge_b1, *edge_w2, *edge_b2;
  const float *ro_w1, *ro_b1, *ro_w2;
  const float *LA[4], *Lw1[4], *Lb1[4], *Lw2[4], *Lb2[4], *LB[4];

  if (lsz[0] == 3392) { // insertion order at top level
    node_w1 = leaf[0]; node_b1 = leaf[1]; node_w2 = leaf[2]; node_b2 = leaf[3];
    edge_w1 = leaf[4]; edge_b1 = leaf[5]; edge_w2 = leaf[6]; edge_b2 = leaf[7];
    bool lay_ins = (lsz[9] == 9216); // layer dict insertion vs sorted
    for (int l = 0; l < 4; ++l) {
      int b = 8 + 6 * l;
      if (lay_ins) { // A, fc_w1, fc_b1, fc_w2, fc_b2, B
        LA[l] = leaf[b]; Lw1[l] = leaf[b + 1]; Lb1[l] = leaf[b + 2];
        Lw2[l] = leaf[b + 3]; Lb2[l] = leaf[b + 4]; LB[l] = leaf[b + 5];
      } else {       // A, B, fc_b1, fc_b2, fc_w1, fc_w2
        LA[l] = leaf[b]; LB[l] = leaf[b + 1]; Lb1[l] = leaf[b + 2];
        Lb2[l] = leaf[b + 3]; Lw1[l] = leaf[b + 4]; Lw2[l] = leaf[b + 5];
      }
    }
    ro_w1 = leaf[32]; ro_b1 = leaf[33]; ro_w2 = leaf[34];
  } else {            // sorted keys everywhere
    edge_b1 = leaf[0]; edge_b2 = leaf[1]; edge_w1 = leaf[2]; edge_w2 = leaf[3];
    for (int l = 0; l < 4; ++l) {
      int b = 4 + 6 * l; // A, B, fc_b1, fc_b2, fc_w1, fc_w2
      LA[l] = leaf[b]; LB[l] = leaf[b + 1]; Lb1[l] = leaf[b + 2];
      Lb2[l] = leaf[b + 3]; Lw1[l] = leaf[b + 4]; Lw2[l] = leaf[b + 5];
    }
    node_b1 = leaf[28]; node_b2 = leaf[29]; node_w1 = leaf[30]; node_w2 = leaf[31];
    ro_b1 = leaf[32]; ro_w1 = leaf[33]; ro_w2 = leaf[34];
  }

  const int* srcI = eidx;
  const int* dstI = eidx + N_EDGES_C;
  const int dims[5] = {32, 96, 160, 192, 192};

  // Workspace carving (floats, 256B-aligned chunks)
  float* wsf = (float*)d_ws;
  size_t off = 0;
  auto carve = [&](size_t n) { float* p = wsf + off; off += (n + 63) & ~(size_t)63; return p; };
  float* temb  = carve((size_t)256 * 32);
  float* hA    = carve((size_t)N_NODES_C * HSTRIDE);
  float* hB    = carve((size_t)N_NODES_C * HSTRIDE);
  float* efeat = carve((size_t)N_EDGES_C * 32);
  float* shb   = carve((size_t)N_EDGES_C * 9);
  float* cnt   = carve((size_t)N_NODES_C);

  // f16 packed-weight arena (256B aligned chunks)
  _Float16* wha = (_Float16*)(wsf + off);
  size_t hoff = 0;
  auto carve16 = [&](size_t n) {
    _Float16* p = wha + hoff; hoff += (n + 127) & ~(size_t)127; return p;
  };
  _Float16* nw1t = carve16((size_t)32 * 128);
  _Float16* nw2t = carve16((size_t)32 * 32);
  _Float16* ew1t = carve16((size_t)32 * 96);
  _Float16* ew2t = carve16((size_t)32 * 32);
  _Float16 *At[4], *fw1t[4], *fw2t[4], *Bt[4];
  for (int l = 0; l < 4; ++l) {
    At[l]   = carve16((size_t)32 * dims[l]);
    fw1t[l] = carve16((size_t)96 * 96);
    fw2t[l] = carve16((size_t)288 * 96);
    Bt[l]   = carve16((size_t)dims[l + 1] * 288);
  }
  _Float16* rw1t = carve16((size_t)96 * 192);
  _Float16* rw2t = carve16((size_t)16 * 96);

  auto pack = [&](const float* W, _Float16* Wt, int K, int N, int Kpad, int Npad) {
    int tot = Npad * Kpad;
    k_pack<<<(tot + 255) / 256, 256, 0, stream>>>(W, Wt, K, N, Kpad, Npad);
  };
  pack(node_w1, nw1t, 106, 32, 128, 32);
  pack(node_w2, nw2t, 32, 32, 32, 32);
  pack(edge_w1, ew1t, 86, 32, 96, 32);
  pack(edge_w2, ew2t, 32, 32, 32, 32);
  for (int l = 0; l < 4; ++l) {
    pack(LA[l],  At[l],   dims[l], 32, dims[l], 32);
    pack(Lw1[l], fw1t[l], 96, 96, 96, 96);
    pack(Lw2[l], fw2t[l], 96, 288, 96, 288);
    pack(LB[l],  Bt[l],   288, dims[l + 1], 288, dims[l + 1]);
  }
  pack(ro_w1, rw1t, 192, 96, 192, 96);
  pack(ro_w2, rw2t, 96, 3, 96, 16);

  hipMemsetAsync(cnt, 0, (size_t)N_NODES_C * sizeof(float), stream);
  k_temb<<<(256 * 32 + 255) / 256, 256, 0, stream>>>(timeI, temb);
  k_counts<<<(N_EDGES_C + 255) / 256, 256, 0, stream>>>(dstI, cnt, N_EDGES_C);
  k_node_mlp<<<N_NODES_C / ME, 64, 0, stream>>>(x, batch, temb, nw1t, node_b1,
                                                nw2t, node_b2, hA);
  k_edge_mlp<<<N_EDGES_C / ME, 64, 0, stream>>>(pos, eattr, srcI, dstI, batch, temb,
                                                ew1t, edge_b1, ew2t, edge_b2,
                                                efeat, shb);

  float* hcur = hA;
  float* hacc = hB;
  for (int l = 0; l < 4; ++l) {
    hipMemsetAsync(hacc, 0, (size_t)N_NODES_C * HSTRIDE * sizeof(float), stream);
    k_layer<<<N_EDGES_C / ME, 128, 0, stream>>>(hcur, efeat, shb, srcI, dstI,
                                                At[l], fw1t[l], Lb1[l], fw2t[l],
                                                Lb2[l], Bt[l], hacc,
                                                dims[l], dims[l + 1]);
    k_finalize<<<(N_NODES_C * dims[l + 1] + 255) / 256, 256, 0, stream>>>(
        hacc, hcur, cnt, dims[l], dims[l + 1], N_NODES_C);
    float* t = hcur; hcur = hacc; hacc = t;
  }

  k_readout<<<N_NODES_C / ME, 128, 0, stream>>>(hcur, rw1t, ro_b1, rw2t,
                                                (float*)d_out);
}
